// SelfMultiheadAttention_82549271429740
// MI455X (gfx1250) — compile-verified, matrix operand order checked by reading, not run on hardware
//
#include <hip/hip_runtime.h>
#include <hip/hip_bf16.h>
#include <math.h>

// ---------------------------------------------------------------------------
// Types for CDNA5 WMMA (wave32): v_wmma_f32_16x16x32_bf16
// ---------------------------------------------------------------------------
typedef __attribute__((ext_vector_type(16))) __bf16 v16bf;
typedef __attribute__((ext_vector_type(8)))  __bf16 v8bf;
typedef __attribute__((ext_vector_type(8)))  float  v8f;

#define EMBED   1024
#define NHEAD   16
#define HDIM    64
#define SEQ     1024
#define BATCH   4
// (HEAD_DIM * 0.1)^-0.5 = 6.4^-0.5
#define QSCALE  0.39528470752104744f

// ---------------------------------------------------------------------------
// CDNA5 async global->LDS copy (ASYNCcnt path). Probe-discovered signature:
//   void __builtin_amdgcn_global_load_async_to_lds_b128(
//       v4i AS(1)* src, v4i AS(3)* dst, imm int offset, imm int cpol)
// Guarded: falls back to a plain VGPR round-trip copy if absent.
// ---------------------------------------------------------------------------
#if defined(__AMDGCN__) && __has_builtin(__builtin_amdgcn_global_load_async_to_lds_b128)
#define ASYNC_LDS 1
#else
#define ASYNC_LDS 0
#endif

typedef int v4i __attribute__((__vector_size__(16)));
typedef __attribute__((address_space(1))) v4i* gv4i_p;
typedef __attribute__((address_space(3))) v4i* sv4i_p;

__device__ __forceinline__ void cp_async_b128(void* lds_dst, const void* gsrc) {
#if ASYNC_LDS
  __builtin_amdgcn_global_load_async_to_lds_b128(
      (gv4i_p)(gsrc), (sv4i_p)(lds_dst), /*imm offset*/ 0, /*cpol*/ 0);
#else
  *(uint4*)lds_dst = *(const uint4*)gsrc;
#endif
}

__device__ __forceinline__ void cp_async_wait() {
#if ASYNC_LDS
#if __has_builtin(__builtin_amdgcn_s_wait_asynccnt)
  __builtin_amdgcn_s_wait_asynccnt(0);
#else
  asm volatile("s_wait_asynccnt 0x0" ::: "memory");
#endif
#endif
}

__device__ __forceinline__ v16bf frag_cat(v8bf lo, v8bf hi) {
  return __builtin_shufflevector(lo, hi, 0,1,2,3,4,5,6,7,8,9,10,11,12,13,14,15);
}

__device__ __forceinline__ v8f wmma_bf16f32(v16bf a, v16bf b, v8f c) {
  // 8 args: (neg_a, A, neg_b, B, c_mod, C, reuse_a, reuse_b)
  return __builtin_amdgcn_wmma_f32_16x16x32_bf16(false, a, false, b, (short)0, c,
                                                 false, false);
}

__device__ __forceinline__ v8f v8f_zero() {
  v8f z = {0.f,0.f,0.f,0.f,0.f,0.f,0.f,0.f};
  return z;
}

// Load a 16x32 bf16 A/B fragment from an LDS (or global) tile whose rows have
// STRIDE bf16 elements and K contiguous.  Per ISA 7.12.2 (16-bit A 16x32):
//   row/col = lane&15 ; lanes 0-15 hold K = k0+[0..7] and k0+[16..23],
//   lanes 16-31 hold K = k0+[8..15] and k0+[24..31].
template <int STRIDE>
__device__ __forceinline__ v16bf load_frag(const __bf16* base, int row, int k0,
                                           int lane) {
  const __bf16* p = base + row * STRIDE + k0 + ((lane >> 4) << 3);
  v8bf lo = *(const v8bf*)p;
  v8bf hi = *(const v8bf*)(p + 16);
  return frag_cat(lo, hi);
}

// ---------------------------------------------------------------------------
// Kernel 1: fused QKV projection.
//   qkv[m, f] = sum_e X[m,e] * Wqkv[f,e] + bqkv[f]        (M=4096, N=3072, K=1024)
//   -> split into q,k,v, q *= QSCALE, store bf16 at [B,H,L,D].
// Block tile 128x128, K-step 32, 8 waves: 4 along M (32 rows) x 2 along N (64 cols).
// (fp32 -> bf16 conversion requires the VGPR round-trip; no async copy here.)
// ---------------------------------------------------------------------------
__global__ __launch_bounds__(256)
void mha_gemm_qkv(const float* __restrict__ X, const float* __restrict__ W,
                  const float* __restrict__ bias,
                  __bf16* __restrict__ qws, __bf16* __restrict__ kws,
                  __bf16* __restrict__ vws) {
  __shared__ __align__(16) __bf16 As[128 * 32];
  __shared__ __align__(16) __bf16 Bs[128 * 32];

  const int tid = threadIdx.x;
  const int lane = tid & 31, w = tid >> 5;
  const int wy = w & 3, wx = w >> 2;
  const int nlo = lane & 15, half = lane >> 4;
  const int m0 = blockIdx.x * 128;
  const int n0 = blockIdx.y * 128;

  v8f c[2][4];
#pragma unroll
  for (int i = 0; i < 2; ++i)
#pragma unroll
    for (int j = 0; j < 4; ++j) c[i][j] = v8f_zero();

  for (int kk = 0; kk < EMBED; kk += 32) {
    // cooperative load + fp32->bf16 convert, A tile [128][32]
#pragma unroll
    for (int i = 0; i < 4; ++i) {
      int idx = tid + 256 * i;               // 1024 float4 chunks
      int r = idx >> 3, c4 = idx & 7;
      float4 f = *(const float4*)(X + (size_t)(m0 + r) * EMBED + kk + c4 * 4);
      __bf16* d = As + r * 32 + c4 * 4;
      d[0] = (__bf16)f.x; d[1] = (__bf16)f.y; d[2] = (__bf16)f.z; d[3] = (__bf16)f.w;
    }
    // B tile [n][k]: Wqkv is (3E, E) row-major, e contiguous -> direct
#pragma unroll
    for (int i = 0; i < 4; ++i) {
      int idx = tid + 256 * i;
      int r = idx >> 3, c4 = idx & 7;
      float4 f = *(const float4*)(W + (size_t)(n0 + r) * EMBED + kk + c4 * 4);
      __bf16* d = Bs + r * 32 + c4 * 4;
      d[0] = (__bf16)f.x; d[1] = (__bf16)f.y; d[2] = (__bf16)f.z; d[3] = (__bf16)f.w;
    }
    __syncthreads();

    v16bf a[2], bfr[4];
#pragma unroll
    for (int i = 0; i < 2; ++i)
      a[i] = load_frag<32>(As, wy * 32 + i * 16 + nlo, 0, lane);
#pragma unroll
    for (int j = 0; j < 4; ++j)
      bfr[j] = load_frag<32>(Bs, wx * 64 + j * 16 + nlo, 0, lane);
#pragma unroll
    for (int i = 0; i < 2; ++i)
#pragma unroll
      for (int j = 0; j < 4; ++j) c[i][j] = wmma_bf16f32(a[i], bfr[j], c[i][j]);
    __syncthreads();
  }

  // epilogue: +bias, Q-scaling, scatter into [B,H,L,D] bf16
#pragma unroll
  for (int i = 0; i < 2; ++i) {
#pragma unroll
    for (int j = 0; j < 4; ++j) {
      int n = n0 + wx * 64 + j * 16 + nlo;
      float bv = bias[n];
      int part = n >> 10;            // 0=q, 1=k, 2=v
      int e = n & 1023, h = e >> 6, d = e & 63;
      __bf16* dst = (part == 0) ? qws : ((part == 1) ? kws : vws);
      float sc = (part == 0) ? QSCALE : 1.0f;
#pragma unroll
      for (int r = 0; r < 8; ++r) {
        int m = m0 + wy * 32 + i * 16 + half * 8 + r;
        int b = m >> 10, l = m & 1023;
        float v = (c[i][j][r] + bv) * sc;
        dst[(((size_t)(b * NHEAD + h) * SEQ) + l) * HDIM + d] = (__bf16)v;
      }
    }
  }
}

// ---------------------------------------------------------------------------
// Kernel 2: fused flash attention for one (b,h) and a 128-row query tile.
//   S = Q K^T + bias (+mask), online softmax, O += P V, O /= l, store bf16
//   at [B, L, E] for the output projection.
// 8 waves, each owns 16 query rows.  Key loop in tiles of 128.
// K tile is staged with CDNA5 async global->LDS copies (pure byte copy);
// V needs a transpose so it round-trips through VGPRs.  attn_bias (268 MB,
// read-once, larger than the 192 MB L2) is streamed with NT hints.
// ---------------------------------------------------------------------------
__global__ __launch_bounds__(256)
void mha_attn_fused(const __bf16* __restrict__ qws, const __bf16* __restrict__ kws,
                    const __bf16* __restrict__ vws, const float* __restrict__ bias,
                    const unsigned char* __restrict__ kpm,
                    __bf16* __restrict__ ows) {
  __shared__ __align__(16) __bf16 Ks[128 * 64];   // [key][d]
  __shared__ __align__(16) __bf16 Vt[64 * 128];   // [d][key]  (transposed)
  __shared__ __align__(16) __bf16 Ps[128 * 128];  // [qrow][key]

  const int tid = threadIdx.x;
  const int lane = tid & 31, w = tid >> 5;
  const int nlo = lane & 15, half = lane >> 4;
  const int bh = blockIdx.y;
  const int b = bh >> 4, h = bh & 15;
  const int q0 = blockIdx.x * 128;

  // Q fragments (16 rows x 64 d -> two 16x32 A fragments), direct from global
  v16bf qa[2];
  {
    const __bf16* qbase = qws + ((size_t)bh * SEQ + q0 + w * 16 + nlo) * HDIM;
#pragma unroll
    for (int s = 0; s < 2; ++s) {
      const __bf16* p = qbase + s * 32 + half * 8;
      qa[s] = frag_cat(*(const v8bf*)p, *(const v8bf*)(p + 16));
    }
  }

  v8f o[4];
#pragma unroll
  for (int df = 0; df < 4; ++df) o[df] = v8f_zero();
  float mrun[8], lrun[8];
#pragma unroll
  for (int r = 0; r < 8; ++r) { mrun[r] = -INFINITY; lrun[r] = 0.f; }

  for (int j = 0; j < SEQ; j += 128) {
    // ---- stage K tile: async global->LDS (no VGPR round-trip) ----
#pragma unroll
    for (int i = 0; i < 4; ++i) {
      int idx = tid + 256 * i;                 // 1024 chunks of 8 bf16
      int key = idx >> 3, c8 = idx & 7;
      cp_async_b128(Ks + key * 64 + c8 * 8,
                    kws + ((size_t)bh * SEQ + j + key) * HDIM + c8 * 8);
    }
    // ---- stage V tile transposed (must pass through VGPRs) ----
#pragma unroll
    for (int i = 0; i < 4; ++i) {
      int idx = tid + 256 * i;
      int key = idx >> 3, c8 = idx & 7;
      v8bf vv = *(const v8bf*)(vws + ((size_t)bh * SEQ + j + key) * HDIM + c8 * 8);
#pragma unroll
      for (int t = 0; t < 8; ++t) Vt[(c8 * 8 + t) * 128 + key] = vv[t];
    }
    cp_async_wait();
    __syncthreads();

    // ---- S = Q K^T  (16 rows x 128 keys, 8 fragments, K-dim = HDIM = 2x32)
    v8f s[8];
#pragma unroll
    for (int nf = 0; nf < 8; ++nf) {
      v8f acc = v8f_zero();
#pragma unroll
      for (int st = 0; st < 2; ++st) {
        v16bf kb = load_frag<64>(Ks, nf * 16 + nlo, st * 32, lane);
        acc = wmma_bf16f32(qa[st], kb, acc);
      }
      s[nf] = acc;
    }

    // ---- + attn_bias (streamed, non-temporal), key padding mask ----
#pragma unroll
    for (int nf = 0; nf < 8; ++nf) {
      int key = j + nf * 16 + nlo;
      bool mk = kpm[b * SEQ + key] != 0;
      const float* bp = bias + ((size_t)bh * SEQ + q0 + w * 16 + half * 8) * SEQ + key;
#pragma unroll
      for (int r = 0; r < 8; ++r) {
        float bv = __builtin_nontemporal_load(bp + (size_t)r * SEQ);
        s[nf][r] = mk ? -INFINITY : (s[nf][r] + bv);
      }
    }

    // ---- online softmax (rows live in half-waves: 16 lanes x 8 vgpr-rows) ----
#pragma unroll
    for (int r = 0; r < 8; ++r) {
      float v = s[0][r];
#pragma unroll
      for (int nf = 1; nf < 8; ++nf) v = fmaxf(v, s[nf][r]);
      v = fmaxf(v, __shfl_xor(v, 1, 32));
      v = fmaxf(v, __shfl_xor(v, 2, 32));
      v = fmaxf(v, __shfl_xor(v, 4, 32));
      v = fmaxf(v, __shfl_xor(v, 8, 32));
      float mnew = fmaxf(mrun[r], v);
      float corr = __expf(mrun[r] - mnew);
      mrun[r] = mnew;
#pragma unroll
      for (int df = 0; df < 4; ++df) o[df][r] *= corr;
      float rsum = 0.f;
#pragma unroll
      for (int nf = 0; nf < 8; ++nf) {
        float p = __expf(s[nf][r] - mnew);
        s[nf][r] = p;
        rsum += p;
      }
      rsum += __shfl_xor(rsum, 1, 32);
      rsum += __shfl_xor(rsum, 2, 32);
      rsum += __shfl_xor(rsum, 4, 32);
      rsum += __shfl_xor(rsum, 8, 32);
      lrun[r] = lrun[r] * corr + rsum;
    }

    // ---- P: C-layout regs -> LDS (bf16) so it can re-enter as A fragments ----
#pragma unroll
    for (int nf = 0; nf < 8; ++nf)
#pragma unroll
      for (int r = 0; r < 8; ++r)
        Ps[(w * 16 + half * 8 + r) * 128 + nf * 16 + nlo] = (__bf16)s[nf][r];
    __syncthreads();

    // ---- O += P V  (K-dim = 128 keys = 4x32; 4 d-tiles of 16) ----
#pragma unroll
    for (int kt = 0; kt < 4; ++kt) {
      v16bf pa = load_frag<128>(Ps + w * 16 * 128, nlo, kt * 32, lane);
#pragma unroll
      for (int df = 0; df < 4; ++df) {
        v16bf vb = load_frag<128>(Vt, df * 16 + nlo, kt * 32, lane);
        o[df] = wmma_bf16f32(pa, vb, o[df]);
      }
    }
    __syncthreads();
  }

  // ---- normalize and store to [B, L, E] bf16 (re-read by GEMM3: keep RT) ----
#pragma unroll
  for (int r = 0; r < 8; ++r) {
    float inv = 1.0f / lrun[r];
    int qrow = q0 + w * 16 + half * 8 + r;
    __bf16* dst = ows + ((size_t)b * SEQ + qrow) * EMBED + h * HDIM;
#pragma unroll
    for (int df = 0; df < 4; ++df)
      dst[df * 16 + nlo] = (__bf16)(o[df][r] * inv);
  }
}

// ---------------------------------------------------------------------------
// Kernel 3: output projection.
//   out[m, f] = sum_e O[m,e] * Wo[f,e] + bo[f]      (M=4096, N=1024, K=1024)
// A is already bf16 (workspace): staged with async global->LDS copies.
// B converted fp32->bf16 through VGPRs.  Result written non-temporally
// (written once, never re-read on device).
// ---------------------------------------------------------------------------
__global__ __launch_bounds__(256)
void mha_gemm_out(const __bf16* __restrict__ Abf, const float* __restrict__ W,
                  const float* __restrict__ bias, float* __restrict__ out) {
  __shared__ __align__(16) __bf16 As[128 * 32];
  __shared__ __align__(16) __bf16 Bs[128 * 32];

  const int tid = threadIdx.x;
  const int lane = tid & 31, w = tid >> 5;
  const int wy = w & 3, wx = w >> 2;
  const int nlo = lane & 15, half = lane >> 4;
  const int m0 = blockIdx.x * 128;
  const int n0 = blockIdx.y * 128;

  v8f c[2][4];
#pragma unroll
  for (int i = 0; i < 2; ++i)
#pragma unroll
    for (int j = 0; j < 4; ++j) c[i][j] = v8f_zero();

  for (int kk = 0; kk < EMBED; kk += 32) {
    // A tile: async byte copy (already bf16)
#pragma unroll
    for (int i = 0; i < 2; ++i) {
      int idx = tid + 256 * i;                 // 512 chunks of 8 bf16
      int r = idx >> 2, c8 = idx & 3;
      cp_async_b128(As + r * 32 + c8 * 8,
                    Abf + (size_t)(m0 + r) * EMBED + kk + c8 * 8);
    }
    // B tile: fp32 -> bf16 convert
#pragma unroll
    for (int i = 0; i < 4; ++i) {
      int idx = tid + 256 * i;
      int r = idx >> 3, c4 = idx & 7;
      float4 f = *(const float4*)(W + (size_t)(n0 + r) * EMBED + kk + c4 * 4);
      __bf16* d = Bs + r * 32 + c4 * 4;
      d[0] = (__bf16)f.x; d[1] = (__bf16)f.y; d[2] = (__bf16)f.z; d[3] = (__bf16)f.w;
    }
    cp_async_wait();
    __syncthreads();

    v16bf a[2], bfr[4];
#pragma unroll
    for (int i = 0; i < 2; ++i)
      a[i] = load_frag<32>(As, wy * 32 + i * 16 + nlo, 0, lane);
#pragma unroll
    for (int j = 0; j < 4; ++j)
      bfr[j] = load_frag<32>(Bs, wx * 64 + j * 16 + nlo, 0, lane);
#pragma unroll
    for (int i = 0; i < 2; ++i)
#pragma unroll
      for (int j = 0; j < 4; ++j) c[i][j] = wmma_bf16f32(a[i], bfr[j], c[i][j]);
    __syncthreads();
  }

#pragma unroll
  for (int i = 0; i < 2; ++i) {
#pragma unroll
    for (int j = 0; j < 4; ++j) {
      int n = n0 + wx * 64 + j * 16 + nlo;
      float bv = bias[n];
#pragma unroll
      for (int r = 0; r < 8; ++r) {
        int m = m0 + wy * 32 + i * 16 + half * 8 + r;
        __builtin_nontemporal_store(c[i][j][r] + bv, &out[(size_t)m * EMBED + n]);
      }
    }
  }
}

// ---------------------------------------------------------------------------
// Launcher
// ---------------------------------------------------------------------------
extern "C" void kernel_launch(void* const* d_in, const int* in_sizes, int n_in,
                              void* d_out, int out_size, void* d_ws, size_t ws_size,
                              hipStream_t stream) {
  const float*         query = (const float*)d_in[0];          // (B, L, E) fp32
  const unsigned char* kpm   = (const unsigned char*)d_in[1];  // (B, L) bool
  const float*         abias = (const float*)d_in[2];          // (B*H, L, L) fp32
  const float*         Wqkv  = (const float*)d_in[3];          // (3E, E) fp32
  const float*         bqkv  = (const float*)d_in[4];          // (3E,)  fp32
  const float*         Wo    = (const float*)d_in[5];          // (E, E) fp32
  const float*         bo    = (const float*)d_in[6];          // (E,)   fp32
  float*               out   = (float*)d_out;                  // (B, L, E) fp32

  // bf16 workspace: q | k | v  at [B,H,L,D]  +  attn output at [B,L,E]
  const size_t per = (size_t)BATCH * NHEAD * SEQ * HDIM;       // 4 Mi elements
  __bf16* qws = (__bf16*)d_ws;
  __bf16* kws = qws + per;
  __bf16* vws = kws + per;
  __bf16* ows = vws + per;

  mha_gemm_qkv<<<dim3(32, 24), 256, 0, stream>>>(query, Wqkv, bqkv, qws, kws, vws);
  mha_attn_fused<<<dim3(8, 64), 256, 0, stream>>>(qws, kws, vws, abias, kpm, ows);
  mha_gemm_out<<<dim3(32, 8), 256, 0, stream>>>(ows, Wo, bo, out);
}
